// MultiHeadAttention_73813307949297
// MI455X (gfx1250) — matrix operand order repaired and checked by reading, not compiled, not run on hardware
//
#include <hip/hip_runtime.h>
#include <hip/hip_bf16.h>
#include <math.h>

// ---------------------------------------------------------------------------
// MHA for MI455X / gfx1250: f16 WMMA (f32 accum), register-blocked GEMMs,
// flash attention with Tensor-Data-Mover K staging, WMMA softmax row-sums.
//   B=2, S=2048, E=1024, H=16, D=64
// ---------------------------------------------------------------------------

#define EMBED 1024
#define HEADS 16
#define HDIM  64
#define BATCH 2
#define SEQL  2048
#define MROWS (BATCH * SEQL)   // 4096

typedef __attribute__((ext_vector_type(16))) _Float16 v16h;
typedef __attribute__((ext_vector_type(8)))  float    v8f;
typedef __attribute__((ext_vector_type(4)))  unsigned int v4u;
typedef __attribute__((ext_vector_type(4)))  int      v4i;
typedef __attribute__((ext_vector_type(8)))  int      v8i;

union ABreg { v16h h; uint32_t u[8]; };

// A-matrix (16x32 f16) K offset for packed pair v (ISA 7.12.2):
//  VGPR v<4 : K = kb + 2v      (kb = 8*(lane>=16))
//  VGPR v>=4: K = 16 + kb + 2(v-4)
__device__ __forceinline__ int a_koff(int lane, int v) {
    int kb = (lane >> 4) * 8;
    return (v < 4) ? (kb + 2 * v) : (16 + kb + 2 * (v - 4));
}

__device__ __forceinline__ v8f wmma_f16(const ABreg& a, const ABreg& b, v8f c) {
    return __builtin_amdgcn_wmma_f32_16x16x32_f16(false, a.h, false, b.h,
                                                  (short)0, c, false, false);
}

// XOR-lane max via ds_swizzle (group-of-32 mode: and=0x1F, or=0, xor=MASK).
template <int MASK>
__device__ __forceinline__ float swz_xor_max(float x) {
    int y = __builtin_amdgcn_ds_swizzle(__float_as_int(x), (MASK << 10) | 0x1F);
    return fmaxf(x, __int_as_float(y));
}

// Multiply a packed f16 pair by 0.125 (folds 1/sqrt(HDIM) into Q).
__device__ __forceinline__ uint32_t pk_scale8(uint32_t x) {
    union { uint32_t u; _Float16 h[2]; } a;
    a.u = x;
    a.h[0] = (_Float16)(a.h[0] * (_Float16)0.125f);
    a.h[1] = (_Float16)(a.h[1] * (_Float16)0.125f);
    return a.u;
}

// TDM: load a 64x64 f16 tile (row stride EMBED) from global into LDS.
// D# layout per CDNA5 ISA:
//  g0: count=1 | lds_addr | global_addr[56:0] | type=2
//  g1: data_size=2B, tensor_dim0=1024, tensor_dim1=2048, tile=64x64,
//      tensor_dim0_stride=1024 elements
__device__ __forceinline__ void tdm_load_k(uint32_t lds_addr, uint64_t gaddr) {
    v4u g0;
    g0.x = 1u;                                                   // count=1
    g0.y = lds_addr;                                             // lds_addr
    g0.z = (unsigned)gaddr;                                      // addr[31:0]
    g0.w = (unsigned)((gaddr >> 32) & 0x01FFFFFFu) | (2u << 30); // addr[56:32]|type=2
    v8i g1;
    g1[0] = 0x00010000;           // workgroup_mask=0, data_size=1 (2 bytes)
    g1[1] = (int)(1024u << 16);   // tensor_dim0[15:0]  = 1024
    g1[2] = (int)(2048u << 16);   // tensor_dim1[15:0]  = 2048
    g1[3] = (int)(64u << 16);     // tile_dim0 = 64
    g1[4] = 64;                   // tile_dim1 = 64, tile_dim2 = 0
    g1[5] = 1024;                 // tensor_dim0_stride = 1024 elements
    g1[6] = 0;
    g1[7] = 0;
    v4i z4 = {0, 0, 0, 0};
#if __has_include(<hip/amd_detail/amd_gfx1250_TDM.h>)
    v8i z8 = {0, 0, 0, 0, 0, 0, 0, 0};
    __builtin_amdgcn_tensor_load_to_lds(g0, g1, z4, z4, z8, 0);
#else
    __builtin_amdgcn_tensor_load_to_lds(g0, g1, z4, z4, 0);
#endif
}

// ---------------------------------------------------------------------------
// Kernel 1: f32 -> f16 convert (vectorized x4)
// ---------------------------------------------------------------------------
__global__ void cvt_f32_f16(const float* __restrict__ src,
                            _Float16* __restrict__ dst, int n4) {
    int i = blockIdx.x * blockDim.x + threadIdx.x;
    if (i >= n4) return;
    float4 f = ((const float4*)src)[i];
    union { _Float16 h[4]; uint2 u; } o;
    o.h[0] = (_Float16)f.x; o.h[1] = (_Float16)f.y;
    o.h[2] = (_Float16)f.z; o.h[3] = (_Float16)f.w;
    ((uint2*)dst)[i] = o.u;
}

// ---------------------------------------------------------------------------
// Kernel 2: transpose+convert all four weights: Wt[w][n][k] = (f16)W[w][k][n]
// ---------------------------------------------------------------------------
__global__ void transpose_w(const float* __restrict__ W0, const float* __restrict__ W1,
                            const float* __restrict__ W2, const float* __restrict__ W3,
                            _Float16* __restrict__ dst) {
    size_t idx = (size_t)blockIdx.x * blockDim.x + threadIdx.x;
    if (idx >= (size_t)4 * EMBED * EMBED) return;
    int w   = (int)(idx / (EMBED * EMBED));
    int rem = (int)(idx % (EMBED * EMBED));
    int k = rem / EMBED, n = rem % EMBED;
    const float* W = (w == 0) ? W0 : (w == 1) ? W1 : (w == 2) ? W2 : W3;
    dst[(size_t)w * EMBED * EMBED + (size_t)n * EMBED + k] = (_Float16)W[rem];
}

// ---------------------------------------------------------------------------
// Kernel 3: WMMA GEMM, 32x64 output per wave (2x4 tiles, 8 WMMA per K-step).
//   out[m,n] = sum_k A[m,k]*Wt[n,k] + bias[n]
//   MODE 0: f16 row-major   MODE 1: f16 -> [B,H,D,S]   MODE 2: f32 row-major
// ---------------------------------------------------------------------------
template <int MODE>
__global__ void __launch_bounds__(256)
gemm_ws(const _Float16* __restrict__ Ah,    // [MROWS, EMBED]
        const _Float16* __restrict__ Bt,    // [EMBED(n), EMBED(k)]
        const float* __restrict__ bias,     // [EMBED]
        void* __restrict__ outp) {
    const int lane = threadIdx.x & 31;
    const int wave = threadIdx.x >> 5;
    const int tile = blockIdx.x * 8 + wave;
    const int TN = EMBED / 64;              // 16 column super-tiles
    const int mt = tile / TN, nt = tile % TN;
    const int m0 = mt * 32, n0 = nt * 64;

    const _Float16* aptr0 = Ah + (size_t)(m0 + (lane & 15)) * EMBED;
    const _Float16* aptr1 = aptr0 + (size_t)16 * EMBED;
    const _Float16* bptr  = Bt + (size_t)(n0 + (lane & 15)) * EMBED;
    const int bkb = (lane >> 4) * 16;       // unused; B uses a_koff order: lanes>=16 hold K=8..15,24..31

    v8f acc[2][4] = {};
    for (int ks = 0; ks < EMBED / 32; ++ks) {
        __builtin_prefetch(aptr0 + ks * 32 + 256, 0, 1);
        __builtin_prefetch(bptr + ks * 32 + 256, 0, 1);
        ABreg a[2], b[4];
#pragma unroll
        for (int v = 0; v < 8; ++v) {
            a[0].u[v] = *(const uint32_t*)(aptr0 + ks * 32 + a_koff(lane, v));
            a[1].u[v] = *(const uint32_t*)(aptr1 + ks * 32 + a_koff(lane, v));
        }
#pragma unroll
        for (int n = 0; n < 4; ++n)
#pragma unroll
            for (int v = 0; v < 8; ++v)
                b[n].u[v] = *(const uint32_t*)(bptr + (size_t)(n * 16) * EMBED +
                                               ks * 32 + a_koff(lane, v));
#pragma unroll
        for (int mi = 0; mi < 2; ++mi)
#pragma unroll
            for (int n = 0; n < 4; ++n)
                acc[mi][n] = wmma_f16(a[mi], b[n], acc[mi][n]);
    }

#pragma unroll
    for (int mi = 0; mi < 2; ++mi) {
        const int rbase = m0 + mi * 16 + (lane >> 4) * 8;
#pragma unroll
        for (int n = 0; n < 4; ++n) {
            const int col = n0 + n * 16 + (lane & 15);
            const float bv = bias[col];
#pragma unroll
            for (int r = 0; r < 8; ++r) {
                float val = acc[mi][n][r] + bv;
                int row = rbase + r;
                if (MODE == 2) {
                    ((float*)outp)[(size_t)row * EMBED + col] = val;
                } else if (MODE == 0) {
                    ((_Float16*)outp)[(size_t)row * EMBED + col] = (_Float16)val;
                } else {
                    int bb = row / SEQL, s = row % SEQL;
                    int hh = col / HDIM, d = col % HDIM;
                    ((_Float16*)outp)[(((size_t)(bb * HEADS + hh)) * HDIM + d) * SEQL + s] =
                        (_Float16)val;
                }
            }
        }
    }
}

// ---------------------------------------------------------------------------
// Kernel 4: RoPE in-place on f16 Q and K. One thread per (m, h, j<32) pair.
// ---------------------------------------------------------------------------
__global__ void rope_qk(_Float16* __restrict__ Q, _Float16* __restrict__ K) {
    int idx = blockIdx.x * blockDim.x + threadIdx.x;  // MROWS*HEADS*32
    int j = idx & 31;
    int t = idx >> 5;
    int h = t & (HEADS - 1);
    int m = t >> 4;
    int s = m & (SEQL - 1);
    float inv = __powf(10000.0f, -(float)(2 * j) * (1.0f / HDIM));
    float sn, c;
    __sincosf((float)s * inv, &sn, &c);
    size_t base = (size_t)m * EMBED + h * HDIM;
    float q1 = (float)Q[base + j], q2 = (float)Q[base + j + 32];
    Q[base + j]      = (_Float16)(q1 * c - q2 * sn);
    Q[base + j + 32] = (_Float16)(q2 * c + q1 * sn);
    float k1 = (float)K[base + j], k2 = (float)K[base + j + 32];
    K[base + j]      = (_Float16)(k1 * c - k2 * sn);
    K[base + j + 32] = (_Float16)(k2 * c + k1 * sn);
}

// ---------------------------------------------------------------------------
// Kernel 5: flash attention. One wave per (b, h, 32-query tile), 4 waves/blk.
// Per 64-key TDM-staged tile, processed as two 32-key softmax rounds to keep
// VGPR pressure under 256:
//   - tensor_load_to_lds stages K (64x64 f16, 8 KB) -> s_wait_tensorcnt
//   - per round: 8 WMMA Q*K^T (LDS K operands shared by both query sub-tiles)
//   - online softmax: scale folded into Q; stage-major ds_swizzle row max;
//     row-sums via P @ ones on the matrix pipe
//   - LDS C->A transpose of P, 8 WMMA P*V + 2 WMMA for the ones column
// ---------------------------------------------------------------------------
__global__ void __launch_bounds__(128)
flash_attn(const _Float16* __restrict__ Q, const _Float16* __restrict__ K,
           const _Float16* __restrict__ Vt,   // [B,H,D,S]
           _Float16* __restrict__ ctx) {      // [MROWS, EMBED]
    __shared__ _Float16 lds_k[4][64 * HDIM];  // 8 KB/wave: TDM-staged K tile
    __shared__ _Float16 lds_p[4][32 * 32];    // 2 KB/wave: P transpose buffer
    const int lane = threadIdx.x & 31;
    const int wave = threadIdx.x >> 5;        // 0..3
    const int flat = blockIdx.x * 4 + wave;
    const int QT = SEQL / 32;                 // 64 query tiles per (b,h)
    const int qt = flat % QT;
    const int bh = flat / QT;                 // 0..B*H-1
    const int b = bh / HEADS, h = bh % HEADS;
    const int q0 = qt * 32;

    _Float16* myk = lds_k[wave];
    _Float16* myp = lds_p[wave];
    const uint32_t myk_lds = (uint32_t)(uintptr_t)myk;  // LDS aperture low bits

    // Preload 32 queries (2 sub-tiles x 2 d-steps), scale 1/8 folded in.
    const _Float16* qbase = Q + (size_t)(b * SEQL + q0) * EMBED + h * HDIM;
    ABreg aq[2][2];
#pragma unroll
    for (int qs = 0; qs < 2; ++qs)
#pragma unroll
        for (int v = 0; v < 8; ++v) {
            const _Float16* qp = qbase + (size_t)(qs * 16 + (lane & 15)) * EMBED;
            aq[qs][0].u[v] = pk_scale8(*(const uint32_t*)(qp + a_koff(lane, v)));
            aq[qs][1].u[v] = pk_scale8(*(const uint32_t*)(qp + 32 + a_koff(lane, v)));
        }

    ABreg ones;
#pragma unroll
    for (int v = 0; v < 8; ++v) ones.u[v] = 0x3C003C00u;  // packed f16 1.0

    v8f acc[2][4] = {};
    v8f acc_l[2] = {};                        // row-sums via P @ ones
    float mrow[2][8];
#pragma unroll
    for (int qs = 0; qs < 2; ++qs)
#pragma unroll
        for (int r = 0; r < 8; ++r) mrow[qs][r] = -1e30f;

    const _Float16* kbase = K + (size_t)(b * SEQL) * EMBED + h * HDIM;
    const _Float16* vbase = Vt + (size_t)bh * HDIM * SEQL;
    const int bkb = (lane >> 4) * 16;

    for (int sk = 0; sk < SEQL; sk += 64) {
        // ---- TDM-stage K tile (64 keys x 64 f16) into LDS ----
        asm volatile("s_wait_dscnt 0x0" ::: "memory");  // prior LDS reads done
        uint64_t gaddr = (uint64_t)(uintptr_t)kbase + (uint64_t)sk * (EMBED * 2);
        tdm_load_k(myk_lds, gaddr);
        __builtin_amdgcn_s_wait_tensorcnt(0);
        asm volatile("" ::: "memory");

        // ---- two 32-key softmax rounds over the staged tile ----
#pragma unroll
        for (int half = 0; half < 2; ++half) {
            const _Float16* khalf = myk + half * 32 * HDIM;
            const int skh = sk + half * 32;

            v8f sc[2][2];
#pragma unroll
            for (int kt = 0; kt < 2; ++kt) {
                ABreg b0, b1;
                const _Float16* kp = khalf + (kt * 16 + (lane & 15)) * HDIM;
#pragma unroll
                for (int v = 0; v < 8; ++v) {
                    b0.u[v] = *(const uint32_t*)(kp + a_koff(lane, v));
                    b1.u[v] = *(const uint32_t*)(kp + 32 + a_koff(lane, v));
                }
#pragma unroll
                for (int qs = 0; qs < 2; ++qs) {
                    v8f z = {};
                    sc[qs][kt] = wmma_f16(aq[qs][0], b0, z);
                    sc[qs][kt] = wmma_f16(aq[qs][1], b1, sc[qs][kt]);
                }
            }

            // online softmax per query sub-tile
#pragma unroll
            for (int qs = 0; qs < 2; ++qs) {
                float mx[8];
#pragma unroll
                for (int r = 0; r < 8; ++r)
                    mx[r] = fmaxf(sc[qs][0][r], sc[qs][1][r]);
#pragma unroll
                for (int r = 0; r < 8; ++r) mx[r] = swz_xor_max<1>(mx[r]);
#pragma unroll
                for (int r = 0; r < 8; ++r) mx[r] = swz_xor_max<2>(mx[r]);
#pragma unroll
                for (int r = 0; r < 8; ++r) mx[r] = swz_xor_max<4>(mx[r]);
#pragma unroll
                for (int r = 0; r < 8; ++r) mx[r] = swz_xor_max<8>(mx[r]);
#pragma unroll
                for (int r = 0; r < 8; ++r) {
                    float mnew = fmaxf(mrow[qs][r], mx[r]);
                    float corr = __expf(mrow[qs][r] - mnew);
                    mrow[qs][r] = mnew;
                    acc_l[qs][r] *= corr;
#pragma unroll
                    for (int nb = 0; nb < 4; ++nb) acc[qs][nb][r] *= corr;
                    int row = qs * 16 + r + 8 * (lane >> 4);
                    myp[row * 32 + (lane & 15)] =
                        (_Float16)__expf(sc[qs][0][r] - mnew);
                    myp[row * 32 + 16 + (lane & 15)] =
                        (_Float16)__expf(sc[qs][1][r] - mnew);
                }
            }
            asm volatile("s_wait_dscnt 0x0" ::: "memory");

            // reload P in A layout (16 queries x 32 keys per sub-tile)
            ABreg pa[2];
#pragma unroll
            for (int qs = 0; qs < 2; ++qs)
#pragma unroll
                for (int v = 0; v < 8; ++v)
                    pa[qs].u[v] = *(const uint32_t*)(myp + (qs * 16 + (lane & 15)) * 32 +
                                                     a_koff(lane, v));

            // row-sums on the matrix pipe: acc_l += P @ ones
#pragma unroll
            for (int qs = 0; qs < 2; ++qs)
                acc_l[qs] = wmma_f16(pa[qs], ones, acc_l[qs]);

            // P*V: V B-tiles loaded once, reused for both query sub-tiles
#pragma unroll
            for (int nb = 0; nb < 4; ++nb) {
                ABreg bv;
                const _Float16* vp =
                    vbase + (size_t)(nb * 16 + (lane & 15)) * SEQL + skh;
#pragma unroll
                for (int v = 0; v < 8; ++v)
                    bv.u[v] = *(const uint32_t*)(vp + a_koff(lane, v));
#pragma unroll
                for (int qs = 0; qs < 2; ++qs)
                    acc[qs][nb] = wmma_f16(pa[qs], bv, acc[qs][nb]);
            }
        }
    }

    // ---- normalize by WMMA-computed row sums and store ctx f16 ----
#pragma unroll
    for (int qs = 0; qs < 2; ++qs)
#pragma unroll
        for (int r = 0; r < 8; ++r) {
            float inv = 1.0f / acc_l[qs][r];
            int row = b * SEQL + q0 + qs * 16 + r + 8 * (lane >> 4);
            size_t ob = (size_t)row * EMBED + h * HDIM + (lane & 15);
            ctx[ob]      = (_Float16)(acc[qs][0][r] * inv);
            ctx[ob + 16] = (_Float16)(acc[qs][1][r] * inv);
            ctx[ob + 32] = (_Float16)(acc[qs][2][r] * inv);
            ctx[ob + 48] = (_Float16)(acc[qs][3][r] * inv);
        }
}

// ---------------------------------------------------------------------------
// Host launcher
// ---------------------------------------------------------------------------
extern "C" void kernel_launch(void* const* d_in, const int* in_sizes, int n_in,
                              void* d_out, int out_size, void* d_ws, size_t ws_size,
                              hipStream_t stream) {
    (void)in_sizes; (void)n_in; (void)out_size; (void)ws_size;
    const float* x  = (const float*)d_in[0];
    const float* Wq = (const float*)d_in[1];
    const float* bq = (const float*)d_in[2];
    const float* Wk = (const float*)d_in[3];
    const float* bk = (const float*)d_in[4];
    const float* Wv = (const float*)d_in[5];
    const float* bv = (const float*)d_in[6];
    const float* Wo = (const float*)d_in[7];
    const float* bo = (const float*)d_in[8];

    char* ws = (char*)d_ws;
    const size_t XH  = (size_t)MROWS * EMBED * sizeof(_Float16);   // 8 MiB
    const size_t WT1 = (size_t)EMBED * EMBED * sizeof(_Float16);   // 2 MiB
    _Float16* xh   = (_Float16*)(ws);
    _Float16* wt   = (_Float16*)(ws + XH);                         // 4 weights
    _Float16* Qh   = (_Float16*)(ws + XH + 4 * WT1);
    _Float16* Kh   = (_Float16*)(ws + 2 * XH + 4 * WT1);
    _Float16* Vth  = (_Float16*)(ws + 3 * XH + 4 * WT1);
    _Float16* ctxh = (_Float16*)(ws + 4 * XH + 4 * WT1);

    // 1) x -> f16
    {
        int n4 = MROWS * EMBED / 4;
        cvt_f32_f16<<<(n4 + 255) / 256, 256, 0, stream>>>(x, xh, n4);
    }
    // 2) weights -> f16 transposed
    {
        size_t n = (size_t)4 * EMBED * EMBED;
        transpose_w<<<(unsigned)((n + 255) / 256), 256, 0, stream>>>(Wq, Wk, Wv, Wo, wt);
    }
    // 3) QKV projections (V stored transposed [B,H,D,S])
    const int gemm_blocks = (MROWS / 32) * (EMBED / 64) / 8;  // 256
    gemm_ws<0><<<gemm_blocks, 256, 0, stream>>>(xh, wt + 0 * (size_t)EMBED * EMBED, bq, Qh);
    gemm_ws<0><<<gemm_blocks, 256, 0, stream>>>(xh, wt + 1 * (size_t)EMBED * EMBED, bk, Kh);
    gemm_ws<1><<<gemm_blocks, 256, 0, stream>>>(xh, wt + 2 * (size_t)EMBED * EMBED, bv, Vth);
    // 4) RoPE on Q, K
    {
        int n = MROWS * HEADS * 32;
        rope_qk<<<n / 256, 256, 0, stream>>>(Qh, Kh);
    }
    // 5) attention (4 waves per block, 40 KB LDS)
    {
        int waves = BATCH * HEADS * (SEQL / 32);  // 2048
        flash_attn<<<waves / 4, 128, 0, stream>>>(Qh, Kh, Vth, ctxh);
    }
    // 6) output projection -> f32 d_out
    gemm_ws<2><<<gemm_blocks, 256, 0, stream>>>(ctxh, wt + 3 * (size_t)EMBED * EMBED, bo,
                                                d_out);
}
